// TemporalFusion_10299331576406
// MI455X (gfx1250) — matrix operand order repaired and checked by reading, not compiled, not run on hardware
//
#include <hip/hip_runtime.h>

#define BSZ 2
#define FRM 4
#define CHN 256
#define GW  128
#define GH  128
#define HWN 16384
#define NHEAD 8
#define NPTS 4
#define NLAYER 3
#define KQ 1024
#define NFFN 1024
#define DHEAD 32

typedef __attribute__((ext_vector_type(16))) __bf16 v16bf;
typedef __attribute__((ext_vector_type(8)))  float  v8f;
typedef __attribute__((ext_vector_type(4)))  unsigned uint4v;
typedef __attribute__((ext_vector_type(8)))  unsigned uint8v;

__device__ __forceinline__ __bf16 f2bf(float f) {
  unsigned u = __float_as_uint(f);
  unsigned r = u + 0x7FFFu + ((u >> 16) & 1u);
  unsigned short s = (unsigned short)(r >> 16);
  __bf16 o; __builtin_memcpy(&o, &s, 2); return o;
}
__device__ __forceinline__ float bf2f(__bf16 h) {
  unsigned short s; __builtin_memcpy(&s, &h, 2);
  unsigned u = ((unsigned)s) << 16;
  return __uint_as_float(u);
}
__device__ __forceinline__ float gelu_tanh(float x) {
  const float k0 = 0.7978845608028654f;
  const float k1 = 0.044715f;
  float t = tanhf(k0 * (x + k1 * x * x * x));
  return 0.5f * x * (1.0f + t);
}

// ---------------- score: max over 10 channels of clipped sigmoid -----------
__global__ void score_kernel(const float* __restrict__ hm, unsigned* __restrict__ sbits) {
  int i = blockIdx.x * blockDim.x + threadIdx.x;
  if (i >= BSZ * HWN) return;
  int b = i >> 14, hw = i & (HWN - 1);
  float sc = 0.0f;
  for (int ch = 0; ch < 10; ++ch) {
    float r = hm[((size_t)b * 10 + ch) * HWN + hw];
    float sg = 1.0f / (1.0f + __expf(-r));
    sg = fminf(fmaxf(sg, 1e-4f), 1.0f - 1e-4f);
    sc = fmaxf(sc, sg);
  }
  sbits[i] = __float_as_uint(sc);   // positive floats: bit order == value order
}

// ---------------- exact top-K via binary search on float bits --------------
__global__ __launch_bounds__(1024)
void topk_kernel(const unsigned* __restrict__ sbits, int* __restrict__ mask_ind) {
  __shared__ unsigned sh_cnt[32];
  __shared__ unsigned sh_tot;
  __shared__ int sh_slot;
  int b = blockIdx.x;
  const unsigned* sb = sbits + (size_t)b * HWN;
  int tid = threadIdx.x;

  unsigned lo = 0u, hi = 0xFFFFFFFFu;   // count_ge(lo) >= K, count_ge(hi) < K
  while (hi - lo > 1u) {
    unsigned mid = lo + ((hi - lo) >> 1);
    int c = 0;
    for (int i = tid; i < HWN; i += 1024) c += (sb[i] >= mid) ? 1 : 0;
    for (int d = 16; d >= 1; d >>= 1) c += __shfl_xor(c, d);
    if ((tid & 31) == 0) sh_cnt[tid >> 5] = (unsigned)c;
    __syncthreads();
    if (tid == 0) {
      unsigned t = 0;
      for (int w = 0; w < 32; ++w) t += sh_cnt[w];
      sh_tot = t;
    }
    __syncthreads();
    unsigned total = sh_tot;
    if (total >= (unsigned)KQ) lo = mid; else hi = mid;
    __syncthreads();
  }
  unsigned V = lo;   // K-th largest value (as bits)

  int c = 0;
  for (int i = tid; i < HWN; i += 1024) c += (sb[i] > V) ? 1 : 0;
  for (int d = 16; d >= 1; d >>= 1) c += __shfl_xor(c, d);
  if ((tid & 31) == 0) sh_cnt[tid >> 5] = (unsigned)c;
  __syncthreads();
  if (tid == 0) {
    unsigned t = 0;
    for (int w = 0; w < 32; ++w) t += sh_cnt[w];
    sh_tot = t;
    sh_slot = 0;
  }
  __syncthreads();
  int c1 = (int)sh_tot;

  for (int i = tid; i < HWN; i += 1024) {
    if (sb[i] > V) {
      int p = atomicAdd(&sh_slot, 1);
      mask_ind[b * KQ + p] = i;
    }
  }
  for (int i = tid; i < HWN; i += 1024) {
    if (sb[i] == V) {
      int rank = 0;
      for (int j = 0; j < i; ++j) rank += (sb[j] == V) ? 1 : 0;
      if (c1 + rank < KQ) mask_ind[b * KQ + c1 + rank] = i;
    }
  }
}

// ---------------- gather preds at selected cells ---------------------------
__global__ void gather_preds_kernel(const float* __restrict__ preds,
                                    const int* __restrict__ mask_ind,
                                    float* __restrict__ qraw) {
  int idx = blockIdx.x * blockDim.x + threadIdx.x;
  if (idx >= BSZ * KQ * 70) return;
  int j = idx % 70, bk = idx / 70, b = bk >> 10;
  int ind = mask_ind[bk];
  qraw[bk * 70 + j] = preds[((size_t)b * 70 + j) * HWN + ind];
}

// ---------------- qn = q + learned positional encoding ---------------------
__global__ void qn_kernel(const float* __restrict__ q, const int* __restrict__ mask_ind,
                          const float* __restrict__ row_embed, const float* __restrict__ col_embed,
                          float* __restrict__ qn) {
  int idx = blockIdx.x * blockDim.x + threadIdx.x;   // BSZ*KQ*CHN
  int cc = idx & (CHN - 1), bk = idx >> 8;
  int ind = mask_ind[bk];
  int w = ind & (GW - 1), h = ind >> 7;
  float pe = (cc < 128) ? col_embed[w * 128 + cc] : row_embed[h * 128 + (cc - 128)];
  qn[idx] = q[idx] + pe;
}

// ---------------- val = transpose(x) + time_pos, f32 -> bf16 ---------------
__global__ __launch_bounds__(256)
void make_val_kernel(const float* __restrict__ x, const float* __restrict__ time_pos,
                     void* __restrict__ valp) {
  __shared__ float tile[64][65];
  __bf16* val = (__bf16*)valp;
  int hw0 = blockIdx.x * 64;
  int c0  = blockIdx.y * 64;
  int bf  = blockIdx.z;          // b*FRM + f
  int f   = bf & (FRM - 1);
  const float* xp = x + ((size_t)bf * CHN) * HWN;
  #pragma unroll
  for (int i = 0; i < 16; ++i) {
    int idx = threadIdx.x + i * 256;
    int ci = idx >> 6, hj = idx & 63;
    tile[ci][hj] = xp[(size_t)(c0 + ci) * HWN + hw0 + hj];
  }
  __syncthreads();
  __bf16* vp2 = val + ((size_t)bf * HWN + hw0) * CHN + c0;
  #pragma unroll
  for (int i = 0; i < 16; ++i) {
    int idx = threadIdx.x + i * 256;
    int hj = idx >> 6, ci = idx & 63;
    vp2[(size_t)hj * CHN + ci] = f2bf(tile[ci][hj] + time_pos[f * CHN + c0 + ci]);
  }
}

#define BM 64
#define BN 64
#define BK 32

// ---- WMMA fragment loaders (ISA 16-bit A / B layouts), stride in elements --
__device__ __forceinline__ v16bf load_afrag(const __bf16* base, int stride, int row16, int lane) {
  v16bf f;
  const __bf16* p = base + (size_t)(row16 + (lane & 15)) * stride;
  int kb = (lane < 16) ? 0 : 8;
  #pragma unroll
  for (int e = 0; e < 8; ++e) f[e] = p[kb + e];
  #pragma unroll
  for (int e = 0; e < 8; ++e) f[8 + e] = p[kb + 16 + e];
  return f;
}
__device__ __forceinline__ v16bf load_bfrag(const __bf16* base, int stride, int col16, int kofs, int lane) {
  v16bf f;
  const __bf16* p = base + (size_t)(col16 + (lane & 15)) * stride + kofs + ((lane < 16) ? 0 : 8);
  #pragma unroll
  for (int e = 0; e < 16; ++e) f[e] = p[e + ((e < 8) ? 0 : 8)];
  return f;
}

// ---------------- generic bf16 WMMA GEMM: C = act(A @ W + bias) ------------
// A: [M,K] f32 or bf16 row-major; W: [K,N] f32 row-major; C: [M,N] f32 or bf16.
template<bool A_BF16, int ACT, bool OUT_BF16>
__global__ __launch_bounds__(256)
void gemm_kernel(const void* __restrict__ Av, const float* __restrict__ W,
                 const float* __restrict__ bias, void* __restrict__ Cv,
                 int M, int N, int K) {
  __shared__ __bf16 As[BM][BK + 2];
  __shared__ __bf16 Bt[BN][BK + 2];   // transposed: [n][k]
  const int tid  = threadIdx.x;
  const int m0   = blockIdx.x * BM;
  const int n0   = blockIdx.y * BN;
  const int wave = tid >> 5;
  const int lane = tid & 31;
  const int wm   = wave & 3;   // 4 x 16-row subtiles
  const int wn   = wave >> 2;  // 2 x 32-col subtiles
  const float* Af = (const float*)Av;
  const __bf16* Ab = (const __bf16*)Av;

  v8f acc0 = {};
  v8f acc1 = {};

  for (int k0 = 0; k0 < K; k0 += BK) {
    #pragma unroll
    for (int i = 0; i < (BM * BK) / 256; ++i) {
      int idx = tid + i * 256;
      int r = idx >> 5, cc = idx & 31;
      int gc = k0 + cc;
      __bf16 a = f2bf(0.0f);
      if (gc < K) {
        if (A_BF16) a = Ab[(size_t)(m0 + r) * K + gc];
        else        a = f2bf(Af[(size_t)(m0 + r) * K + gc]);
      }
      As[r][cc] = a;
    }
    #pragma unroll
    for (int i = 0; i < (BK * BN) / 256; ++i) {
      int idx = tid + i * 256;
      int r = idx >> 6, cc = idx & 63;   // r = k, cc = n
      int gk = k0 + r;
      float bv = (gk < K) ? W[(size_t)gk * N + (n0 + cc)] : 0.0f;
      Bt[cc][r] = f2bf(bv);
    }
    if (k0 + BK < K) {
      const char* pa = A_BF16
        ? (const char*)&Ab[(size_t)(m0 + (tid & 63)) * K + (k0 + BK)]
        : (const char*)&Af[(size_t)(m0 + (tid & 63)) * K + (k0 + BK)];
      __builtin_prefetch(pa, 0, 1);
    }
    __syncthreads();

    v16bf afrag  = load_afrag(&As[0][0], BK + 2, wm << 4, lane);
    v16bf bfrag0 = load_bfrag(&Bt[0][0], BK + 2, (wn << 5), 0, lane);
    v16bf bfrag1 = load_bfrag(&Bt[0][0], BK + 2, (wn << 5) + 16, 0, lane);
    acc0 = __builtin_amdgcn_wmma_f32_16x16x32_bf16(false, afrag, false, bfrag0,
                                                   (short)0, acc0, false, false);
    acc1 = __builtin_amdgcn_wmma_f32_16x16x32_bf16(false, afrag, false, bfrag1,
                                                   (short)0, acc1, false, false);
    __syncthreads();
  }

  float* Cf = (float*)Cv;
  __bf16* Cb = (__bf16*)Cv;
  #pragma unroll
  for (int r = 0; r < 8; ++r) {
    int row = m0 + (wm << 4) + r + ((lane < 16) ? 0 : 8);
    int col = n0 + (wn << 5) + (lane & 15);
    if (row < M) {
      float v0 = acc0[r] + bias[col];
      float v1 = acc1[r] + bias[col + 16];
      if (ACT == 1) { v0 = gelu_tanh(v0); v1 = gelu_tanh(v1); }
      else if (ACT == 2) { v0 = fmaxf(v0, 0.0f); v1 = fmaxf(v1, 0.0f); }
      if (OUT_BF16) {
        Cb[(size_t)row * N + col]      = f2bf(v0);
        Cb[(size_t)row * N + col + 16] = f2bf(v1);
      } else {
        Cf[(size_t)row * N + col]      = v0;
        Cf[(size_t)row * N + col + 16] = v1;
      }
    }
  }
}

// ---------------- value-projection GEMM with TDM async A-tiles -------------
// A: bf16 [M,256] (val), W: f32 [256,256], C: bf16 [M,256].
// A-tiles (64x32 bf16) are DMA'd global->LDS by the Tensor Data Mover with
// hardware row padding (pad 1 dword per 16 dwords -> row stride 34 bf16),
// double-buffered against the WMMAs; B panel (256x64) staged once.
__global__ __launch_bounds__(256)
void gemm_val_kernel(const void* __restrict__ Av, const float* __restrict__ W,
                     const float* __restrict__ bias, void* __restrict__ Cv, int M) {
  constexpr int KTOT = 256;
  constexpr int NKT = KTOT / BK;            // 8 k-steps
  __shared__ __bf16 As[2][BM][BK + 2];      // TDM pads rows to 34 elements
  __shared__ __bf16 Bt[BN][KTOT + 2];       // [n][k]
  const __bf16* Ab = (const __bf16*)Av;
  const int tid  = threadIdx.x;
  const int m0   = blockIdx.x * BM;
  const int n0   = blockIdx.y * BN;
  const int wave = tid >> 5;
  const int lane = tid & 31;
  const int wm   = wave & 3;
  const int wn   = wave >> 2;

  // D# group1: data_size=2B, pad_enable, pad_interval=16dw, pad_amount=1dw,
  // tensor_dim0=256, tensor_dim1=M(=0x20000), tile=32x64, dim0_stride=256.
  const unsigned G1_W0 = 0x00D10000u;
  const unsigned G1_W1 = 0x01000000u;                 // tensor_dim0 low16 << 16
  const unsigned G1_W3 = (32u << 16) | (unsigned)(M >> 16);
  const unsigned G1_W2 = ((unsigned)M & 0xFFFFu) << 16;

  auto tdm_issue = [&](int kt, int buf) {
    unsigned lds_off = (unsigned)(unsigned long long)(&As[buf][0][0]);
    unsigned long long ga =
        (unsigned long long)(const void*)(Ab + (size_t)m0 * KTOT + kt * BK);
    uint4v g0;
    g0[0] = __builtin_amdgcn_readfirstlane(1u);
    g0[1] = __builtin_amdgcn_readfirstlane(lds_off);
    g0[2] = __builtin_amdgcn_readfirstlane((unsigned)ga);
    g0[3] = __builtin_amdgcn_readfirstlane((unsigned)(ga >> 32) | 0x80000000u);
    uint8v g1;
    g1[0] = __builtin_amdgcn_readfirstlane(G1_W0);
    g1[1] = __builtin_amdgcn_readfirstlane(G1_W1);
    g1[2] = __builtin_amdgcn_readfirstlane(G1_W2);
    g1[3] = __builtin_amdgcn_readfirstlane(G1_W3);
    g1[4] = __builtin_amdgcn_readfirstlane(64u);      // tile_dim1
    g1[5] = __builtin_amdgcn_readfirstlane(256u);     // tensor_dim0_stride
    g1[6] = 0u; g1[7] = 0u;
    uint4v gz; gz[0] = 0u; gz[1] = 0u; gz[2] = 0u; gz[3] = 0u;
    asm volatile("tensor_load_to_lds %0, %1, %2, %3"
                 :: "s"(g0), "s"(g1), "s"(gz), "s"(gz)
                 : "memory");
  };

  if (wave == 0) tdm_issue(0, 0);   // kick off first A tile

  // stage B panel 256x64 -> bf16 transposed [n][k]
  #pragma unroll 4
  for (int i = 0; i < (KTOT * BN) / 256; ++i) {
    int idx = tid + i * 256;
    int kk = idx >> 6, nn = idx & 63;
    Bt[nn][kk] = f2bf(W[(size_t)kk * KTOT + n0 + nn]);
  }

  v8f acc0 = {};
  v8f acc1 = {};

  for (int kt = 0; kt < NKT; ++kt) {
    if (wave == 0) {
      if (kt + 1 < NKT) {
        tdm_issue(kt + 1, (kt + 1) & 1);
        __builtin_amdgcn_s_wait_tensorcnt(1);  // tile kt landed
      } else {
        __builtin_amdgcn_s_wait_tensorcnt(0);
      }
    }
    __syncthreads();

    v16bf afrag  = load_afrag(&As[kt & 1][0][0], BK + 2, wm << 4, lane);
    v16bf bfrag0 = load_bfrag(&Bt[0][0], KTOT + 2, (wn << 5), kt * BK, lane);
    v16bf bfrag1 = load_bfrag(&Bt[0][0], KTOT + 2, (wn << 5) + 16, kt * BK, lane);
    acc0 = __builtin_amdgcn_wmma_f32_16x16x32_bf16(false, afrag, false, bfrag0,
                                                   (short)0, acc0, false, false);
    acc1 = __builtin_amdgcn_wmma_f32_16x16x32_bf16(false, afrag, false, bfrag1,
                                                   (short)0, acc1, false, false);
    __syncthreads();   // before buffer (kt&1) is overwritten at kt+2
  }

  __bf16* Cb = (__bf16*)Cv;
  #pragma unroll
  for (int r = 0; r < 8; ++r) {
    int row = m0 + (wm << 4) + r + ((lane < 16) ? 0 : 8);
    int col = n0 + (wn << 5) + (lane & 15);
    Cb[(size_t)row * KTOT + col]      = f2bf(acc0[r] + bias[col]);
    Cb[(size_t)row * KTOT + col + 16] = f2bf(acc1[r] + bias[col + 16]);
  }
}

// ---------------- deformable sampling: wave per (b,k,head) -----------------
__global__ __launch_bounds__(256)
void deform_attn_kernel(const void* __restrict__ vptr,
                        const float* __restrict__ off_raw,  // [BSZ*KQ, 256]
                        const float* __restrict__ aw_raw,   // [BSZ*KQ, 128]
                        const int* __restrict__ mask_ind,
                        float* __restrict__ attn) {         // [BSZ*KQ, 256]
  const __bf16* v = (const __bf16*)vptr;
  int bk = blockIdx.x;
  int b = bk >> 10;
  int h = threadIdx.x >> 5;
  int lane = threadIdx.x & 31;
  int ind = mask_ind[bk];
  float rx = (float)(ind & (GW - 1));
  float ry = (float)(ind >> 7);

  float a = -1e30f;
  if (lane < 16) a = aw_raw[(size_t)bk * 128 + h * 16 + lane];
  float m = a;
  for (int d = 8; d >= 1; d >>= 1) m = fmaxf(m, __shfl_xor(m, d));
  float ex = (lane < 16) ? __expf(a - m) : 0.0f;
  float sum = ex;
  for (int d = 8; d >= 1; d >>= 1) sum += __shfl_xor(sum, d);
  float w = ex / sum;

  int s16 = lane & 15;
  int cxy = lane >> 4;
  float offv = off_raw[(size_t)bk * 256 + (h * 16 + s16) * 2 + cxy];

  const __bf16* vb = v + (size_t)b * FRM * HWN * CHN + (h << 5) + lane;
  float acc = 0.0f;
  for (int s = 0; s < 16; ++s) {
    int f = s >> 2;
    float aws = __shfl(w, s);
    float ox = __shfl(offv, s);
    float oy = __shfl(offv, s + 16);
    float xs = rx + ox;
    float ys = ry + oy;
    float x0f = floorf(xs), y0f = floorf(ys);
    int x0 = (int)x0f, y0 = (int)y0f;
    float wx1 = xs - x0f, wy1 = ys - y0f;
    float wx0 = 1.0f - wx1, wy0 = 1.0f - wy1;
    const __bf16* vf = vb + (size_t)f * HWN * CHN;
    auto g = [&](int yy, int xx) -> float {
      if (yy < 0 || yy >= GH || xx < 0 || xx >= GW) return 0.0f;
      return bf2f(vf[(size_t)(yy * GW + xx) * CHN]);
    };
    float sval = g(y0, x0) * (wx0 * wy0) + g(y0, x0 + 1) * (wx1 * wy0)
               + g(y0 + 1, x0) * (wx0 * wy1) + g(y0 + 1, x0 + 1) * (wx1 * wy1);
    acc += aws * sval;
  }
  attn[(size_t)bk * 256 + (h << 5) + lane] = acc;
}

// ---------------- residual add + LayerNorm (wave per row) ------------------
__global__ __launch_bounds__(256)
void add_ln_kernel(float* __restrict__ q, const float* __restrict__ t,
                   const float* __restrict__ g, const float* __restrict__ b) {
  int row = blockIdx.x * 8 + (threadIdx.x >> 5);
  int lane = threadIdx.x & 31;
  float v[8];
  float s = 0.0f;
  #pragma unroll
  for (int i = 0; i < 8; ++i) {
    int c = lane + i * 32;
    v[i] = q[(size_t)row * CHN + c] + t[(size_t)row * CHN + c];
    s += v[i];
  }
  for (int d = 16; d >= 1; d >>= 1) s += __shfl_xor(s, d);
  float mean = s * (1.0f / CHN);
  float vs = 0.0f;
  #pragma unroll
  for (int i = 0; i < 8; ++i) { float dv = v[i] - mean; vs += dv * dv; }
  for (int d = 16; d >= 1; d >>= 1) vs += __shfl_xor(vs, d);
  float rstd = rsqrtf(vs * (1.0f / CHN) + 1e-5f);
  #pragma unroll
  for (int i = 0; i < 8; ++i) {
    int c = lane + i * 32;
    q[(size_t)row * CHN + c] = (v[i] - mean) * rstd * g[c] + b[c];
  }
}

// ---------------- scatter back to dense channel-major grid -----------------
__global__ void scatter_kernel(const float* __restrict__ q, const int* __restrict__ mask_ind,
                               float* __restrict__ out) {
  int idx = blockIdx.x * blockDim.x + threadIdx.x;   // BSZ*KQ*CHN
  int c = idx & (CHN - 1), bk = idx >> 8, b = bk >> 10;
  int ind = mask_ind[bk];
  out[((size_t)b * CHN + c) * HWN + ind] = q[idx];
}

// ---------------- host side ------------------------------------------------
template<bool AB, int ACT, bool OB>
static void launch_gemm(const void* A, const float* W, const float* bias, void* C,
                        int M, int N, int K, hipStream_t s) {
  dim3 grid(M / BM, N / BN);
  gemm_kernel<AB, ACT, OB><<<grid, 256, 0, s>>>(A, W, bias, C, M, N, K);
}

extern "C" void kernel_launch(void* const* d_in, const int* in_sizes, int n_in,
                              void* d_out, int out_size, void* d_ws, size_t ws_size,
                              hipStream_t stream) {
  (void)in_sizes; (void)n_in; (void)ws_size;
  const float* x         = (const float*)d_in[0];
  const float* preds     = (const float*)d_in[1];
  const float* hm        = (const float*)d_in[2];
  const float* mlp_W1    = (const float*)d_in[3];
  const float* mlp_b1    = (const float*)d_in[4];
  const float* mlp_W2    = (const float*)d_in[5];
  const float* mlp_b2    = (const float*)d_in[6];
  const float* time_pos  = (const float*)d_in[7];
  const float* row_embed = (const float*)d_in[8];
  const float* col_embed = (const float*)d_in[9];
  const float* so_W      = (const float*)d_in[10];
  const float* so_b      = (const float*)d_in[11];
  const float* aw_W      = (const float*)d_in[12];
  const float* aw_b      = (const float*)d_in[13];
  const float* vp_W      = (const float*)d_in[14];
  const float* vp_b      = (const float*)d_in[15];
  const float* op_W      = (const float*)d_in[16];
  const float* op_b      = (const float*)d_in[17];
  const float* ln1_g     = (const float*)d_in[18];
  const float* ln1_b     = (const float*)d_in[19];
  const float* ffn_W1    = (const float*)d_in[20];
  const float* ffn_b1    = (const float*)d_in[21];
  const float* ffn_W2    = (const float*)d_in[22];
  const float* ffn_b2    = (const float*)d_in[23];
  const float* ln2_g     = (const float*)d_in[24];
  const float* ln2_b     = (const float*)d_in[25];
  float* out = (float*)d_out;
  char* ws = (char*)d_ws;

  size_t off = 0;
  auto take = [&](size_t bytes) -> char* {
    char* p = ws + off;
    off += (bytes + 255) & ~(size_t)255;
    return p;
  };
  unsigned* sbits  = (unsigned*)take((size_t)BSZ * HWN * 4);
  int* mask_ind    = (int*)take((size_t)BSZ * KQ * 4);
  float* qraw      = (float*)take((size_t)BSZ * KQ * 70 * 4);
  float* q         = (float*)take((size_t)BSZ * KQ * CHN * 4);
  float* qn        = (float*)take((size_t)BSZ * KQ * CHN * 4);
  float* hbig      = (float*)take((size_t)BSZ * KQ * NFFN * 4);
  float* tmp       = (float*)take((size_t)BSZ * KQ * CHN * 4);
  float* off_raw   = (float*)take((size_t)BSZ * KQ * 256 * 4);
  float* aw_raw    = (float*)take((size_t)BSZ * KQ * 128 * 4);
  float* attn      = (float*)take((size_t)BSZ * KQ * CHN * 4);
  void* val        = (void*)take((size_t)BSZ * FRM * HWN * CHN * 2);  // bf16
  void* vproj      = (void*)take((size_t)BSZ * FRM * HWN * CHN * 2);  // bf16

  const int NROWS = BSZ * KQ;        // 2048
  const int MVAL  = BSZ * FRM * HWN; // 131072

  // 1) scores + exact top-K per batch
  score_kernel<<<(BSZ * HWN + 255) / 256, 256, 0, stream>>>(hm, sbits);
  topk_kernel<<<BSZ, 1024, 0, stream>>>(sbits, mask_ind);

  // 2) q = gelu(gather(preds) @ W1 + b1) @ W2 + b2
  gather_preds_kernel<<<(NROWS * 70 + 255) / 256, 256, 0, stream>>>(preds, mask_ind, qraw);
  launch_gemm<false, 1, false>(qraw, mlp_W1, mlp_b1, hbig, NROWS, CHN, 70, stream);
  launch_gemm<false, 0, false>(hbig, mlp_W2, mlp_b2, q, NROWS, CHN, CHN, stream);

  // 3) val (bf16, transposed, +time_pos) — built once, L2-resident for all layers
  make_val_kernel<<<dim3(HWN / 64, CHN / 64, BSZ * FRM), 256, 0, stream>>>(x, time_pos, val);

  for (int l = 0; l < NLAYER; ++l) {
    qn_kernel<<<NROWS, 256, 0, stream>>>(q, mask_ind, row_embed, col_embed, qn);
    launch_gemm<false, 0, false>(qn, so_W + (size_t)l * CHN * 256, so_b + l * 256,
                                 off_raw, NROWS, 256, CHN, stream);
    launch_gemm<false, 0, false>(qn, aw_W + (size_t)l * CHN * 128, aw_b + l * 128,
                                 aw_raw, NROWS, 128, CHN, stream);
    // value projection: the big GEMM (131072 x 256 x 256), TDM-fed, bf16 in/out
    gemm_val_kernel<<<dim3(MVAL / BM, CHN / BN), 256, 0, stream>>>(
        val, vp_W + (size_t)l * CHN * CHN, vp_b + l * CHN, vproj, MVAL);
    deform_attn_kernel<<<NROWS, 256, 0, stream>>>(vproj, off_raw, aw_raw, mask_ind, attn);
    launch_gemm<false, 0, false>(attn, op_W + (size_t)l * CHN * CHN, op_b + l * CHN,
                                 tmp, NROWS, CHN, CHN, stream);
    add_ln_kernel<<<NROWS / 8, 256, 0, stream>>>(q, tmp, ln1_g + l * CHN, ln1_b + l * CHN);
    launch_gemm<false, 2, false>(q, ffn_W1 + (size_t)l * CHN * NFFN, ffn_b1 + l * NFFN,
                                 hbig, NROWS, NFFN, CHN, stream);
    launch_gemm<false, 0, false>(hbig, ffn_W2 + (size_t)l * NFFN * CHN, ffn_b2 + l * CHN,
                                 tmp, NROWS, CHN, NFFN, stream);
    add_ln_kernel<<<NROWS / 8, 256, 0, stream>>>(q, tmp, ln2_g + l * CHN, ln2_b + l * CHN);
  }

  // 4) dense scatter
  hipMemsetAsync(out, 0, (size_t)out_size * sizeof(float), stream);
  scatter_kernel<<<NROWS, 256, 0, stream>>>(q, mask_ind, out);
}
